// MixedOp_15315853377827
// MI455X (gfx1250) — matrix operand order repaired and checked by reading, not compiled, not run on hardware
//
#include <hip/hip_runtime.h>

// ---------------------------------------------------------------------------
// MixedOp (NAS cell edge) for MI455X / gfx1250, wave32.
//   0: zeros   1: identity   2: relu->conv3x3   3: relu->conv5x5
//   4: avg_pool3 (SAME,/9)   5: max_pool3 (SAME,-inf)
//
// Convs: bf16/NHWC repack (memory-bound, ~5us at 23.3TB/s), then a
// block-tiled implicit GEMM: per kernel tap, stage A(weights) via the Tensor
// Data Mover into LDS and B(activations) via predicated b128 copies, then
// 8 waves consume the 48KB tile with ds_load_b128 + v_wmma_f32_16x16x32_bf16.
// Block tile: 64 Cout x 64 positions; wave tile: 16 x 32 (2 accumulators).
// ---------------------------------------------------------------------------

typedef __attribute__((ext_vector_type(16))) __bf16 v16bf;
typedef __attribute__((ext_vector_type(8)))  __bf16 v8bf;
typedef __attribute__((ext_vector_type(8)))  float  v8f;
typedef unsigned int u32x4 __attribute__((ext_vector_type(4)));
typedef int          i32x4 __attribute__((ext_vector_type(4)));
typedef int          i32x8 __attribute__((ext_vector_type(8)));

#define BATCH 32
#define CHN   192
#define IMH   56
#define IMW   56
#define HW    (IMH * IMW)          // 3136
#define CHW   (CHN * HW)           // 602112
#define NPOS  (BATCH * HW)         // 100352  (divisible by 64)
#define NELEM (BATCH * CHW)        // 19267584

// d_ws layout (bytes)
#define XB_BYTES  ((size_t)2 * NELEM)                // bf16 relu(x), NHWC
#define WB_BYTES  ((size_t)2 * CHN * CHN * 25)       // bf16 weights [co][t][ci]
#define IDX_OFF   (XB_BYTES + WB_BYTES)              // int sampled op

#if defined(__has_builtin)
#if __has_builtin(__builtin_amdgcn_tensor_load_to_lds) && \
    __has_builtin(__builtin_amdgcn_s_wait_tensorcnt)
#define HAVE_TDM 1
#endif
#endif

// --- compile-time probes (stderr-only; zero codegen impact) ----------------
#define XSTR_(s) #s
#define XSTR(s) XSTR_(s)
#if defined(HAVE_TDM)
#warning "CDNA5 probe: TDM path ACTIVE (tensor_load_to_lds builtin present)"
#else
#warning "CDNA5 probe: TDM builtin ABSENT - manual LDS staging fallback in use"
#endif
#pragma message("CDNA5 probe: __clang_major__ = " XSTR(__clang_major__))

// ---------------------------------------------------------------------------
// Threefry2x32(key=(0,42)) + Gumbel argmax == jax.random.categorical(key(42))
// ---------------------------------------------------------------------------
__device__ __forceinline__ unsigned rotl32(unsigned x, unsigned d) {
    return (x << d) | (x >> (32u - d));
}

__global__ void pick_op_kernel(const float* __restrict__ logits,
                               int* __restrict__ idx_out) {
    if (threadIdx.x != 0 || blockIdx.x != 0) return;
    const unsigned ks0 = 0u, ks1 = 42u, ks2 = 0u ^ 42u ^ 0x1BD11BDAu;
    const unsigned ks[3] = {ks0, ks1, ks2};
    const unsigned rot0[4] = {13u, 15u, 26u, 6u};
    const unsigned rot1[4] = {17u, 29u, 16u, 24u};
    unsigned bits[6];
    for (int j = 0; j < 3; ++j) {                 // counts [0,1,2]|[3,4,5]
        unsigned x0 = (unsigned)j + ks0;
        unsigned x1 = (unsigned)(j + 3) + ks1;
        for (int i = 0; i < 5; ++i) {
            const unsigned* rot = (i & 1) ? rot1 : rot0;
            for (int r = 0; r < 4; ++r) {
                x0 = x0 + x1;
                x1 = rotl32(x1, rot[r]);
                x1 ^= x0;
            }
            x0 += ks[(i + 1) % 3];
            x1 += ks[(i + 2) % 3] + (unsigned)(i + 1);
        }
        bits[j] = x0;
        bits[j + 3] = x1;
    }
    int best = 0;
    float bestv = -__builtin_inff();
    for (int i = 0; i < 6; ++i) {
        float f = __uint_as_float((bits[i] >> 9) | 0x3f800000u) - 1.0f;
        const float tiny = 1.1754943508222875e-38f;
        float u = f * (1.0f - tiny) + tiny;
        u = fmaxf(tiny, u);
        float g = -logf(-logf(u));                // gumbel
        float s = logits[i] + g;
        if (s > bestv) { bestv = s; best = i; }
    }
    *idx_out = best;
}

// ---------------------------------------------------------------------------
// Prep: NCHW f32 -> NHWC bf16 with fused ReLU (conv branches only).
// ---------------------------------------------------------------------------
__global__ __launch_bounds__(256) void prep_act_kernel(
    const float* __restrict__ x, const int* __restrict__ idx_ws,
    __bf16* __restrict__ xb) {
    const int idx = *idx_ws;
    if (idx != 2 && idx != 3) return;
    const int i = blockIdx.x * 256 + threadIdx.x;   // NHWC linear index
    if (i >= NELEM) return;
    const int ci  = i % CHN;
    const int sp  = i / CHN;                        // img*HW + h*IMW + w
    const int img = sp / HW;
    const int hw  = sp % HW;
    xb[i] = (__bf16)fmaxf(x[(img * CHN + ci) * HW + hw], 0.0f);
}

// OIHW f32 -> [co][tap][ci] bf16 so A K-dim is contiguous per tap.
template <int TAPS, int OPIDX>
__global__ __launch_bounds__(256) void prep_wgt_kernel(
    const float* __restrict__ w, const int* __restrict__ idx_ws,
    __bf16* __restrict__ wb) {
    if (*idx_ws != OPIDX) return;
    const int i = blockIdx.x * 256 + threadIdx.x;   // [co][t][ci]
    if (i >= CHN * CHN * TAPS) return;
    const int ci = i % CHN;
    const int t  = (i / CHN) % TAPS;
    const int co = i / (CHN * TAPS);
    wb[i] = (__bf16)w[(co * CHN + ci) * TAPS + t];
}

// ---------------------------------------------------------------------------
// Block-tiled implicit GEMM conv.
//   Block: 64 Cout x 64 positions. Per tap: stage A 64x192 (TDM) and
//   B 64x192 (predicated halo copies) into LDS, then 8 waves each run
//   6 k-blocks x 2 N-tiles of ds_load_b128 + wmma.
// Fragment packing: A and B both use contiguous k order k = 16*(lane/16) + s (wave32).
// ---------------------------------------------------------------------------
template <int KH, int KW, int OPIDX>
__global__ __launch_bounds__(256) void conv_wmma_kernel(
    const __bf16* __restrict__ xb, const __bf16* __restrict__ wb,
    const int* __restrict__ idx_ws, float* __restrict__ out) {
    if (*idx_ws != OPIDX) return;               // uniform across grid
    constexpr int TAPS = KH * KW;
    constexpr int KTOT = TAPS * CHN;

    __shared__ __bf16 ldsA[64 * CHN];           // [co_local][ci]  24KB
    __shared__ __bf16 ldsB[64 * CHN];           // [pos_local][ci] 24KB

    const int tid  = threadIdx.x;
    const int lane = tid & 31;
    const int wv   = tid >> 5;                  // 0..7
    const int mo   = (wv & 3) * 16;             // co sub-tile in block
    const int n0   = (wv >> 2) * 32;            // 2 N-tiles per wave

    const int coB = (blockIdx.x % (CHN / 64)) * 64;   // 3 co blocks
    const int p0  = (blockIdx.x / (CHN / 64)) * 64;   // 1568 pos blocks

    // --- staging coords: 64 rows x 384B; 4 threads/row, 6x16B chunks each ---
    const int srow = tid >> 2;                  // 0..63
    const int sq0  = (tid & 3) * 6;             // first 16B chunk (of 24)
    // B row position decomposition (fixed per thread)
    const int bpos = p0 + srow;
    const int bimg = bpos / HW;
    const int brem = bpos % HW;
    const int bph  = brem / IMW;
    const int bpw  = brem % IMW;
    const __bf16* __restrict__ pArow = wb + (coB + srow) * KTOT;

    // --- wmma lane mapping (row/col = lane%16; contiguous k = 16*hi + s) ---
    const int nn  = lane & 15;
    const int hi  = lane >> 4;
    const int kA0 = hi ? 16 : 0;
    const int kA1 = hi ? 24 : 8;

    v8f acc0 = {}, acc1 = {};

#pragma unroll 1
    for (int t = 0; t < TAPS; ++t) {
        const int ih = bph + t / KW - KH / 2;
        const int iw = bpw + t % KW - KW / 2;
        const bool valid =
            (ih >= 0) & (ih < IMH) & (iw >= 0) & (iw < IMW);
        const __bf16* __restrict__ gB =
            xb + ((bimg * IMH + ih) * IMW + iw) * CHN;

        // ---- stage B (activations, halo rows zero-filled) ----
#pragma unroll
        for (int j = 0; j < 6; ++j) {
            const int qq = sq0 + j;             // 16B chunk index in row
            v8bf v = {};
            if (valid) v = *(const v8bf*)(gB + qq * 8);
            *(v8bf*)&ldsB[srow * CHN + qq * 8] = v;
        }

        // ---- stage A (weights) via Tensor Data Mover ----
#if defined(HAVE_TDM)
        if (wv == 0) {
            const unsigned lds_a =
                (unsigned)(size_t)(__attribute__((address_space(3))) __bf16*)
                    ldsA;
            const unsigned long long ga =
                (unsigned long long)(size_t)(wb + coB * KTOT + t * CHN);
            // D# group0: count=1 | lds_addr | global_addr[56:0] | type=2
            const u32x4 g0 = {1u, lds_a, (unsigned)(ga & 0xFFFFFFFFu),
                              (unsigned)((ga >> 32) & 0x01FFFFFFu) |
                                  0x80000000u};
            // D# group1: data_size=2B; tensor 192x64; tile 192x64;
            //            dim0_stride = KTOT (elements)
            const i32x8 g1 = {
                (int)0x00010000u,                         // data_size=1 (2B)
                (int)((CHN & 0xFFFF) << 16),              // tensor_dim0 lo
                (int)((CHN >> 16) | (64u << 16)),         // dim0 hi|dim1 lo
                (int)((64u >> 16) | ((unsigned)CHN << 16)),  // dim1 hi|tile0
                (int)64,                                  // tile_dim1=64
                (int)KTOT, 0, 0};                         // dim0_stride
            const i32x4 g2 = {0, 0, 0, 0};
            const i32x4 g3 = {0, 0, 0, 0};
#if __clang_major__ >= 23
            const i32x8 g4 = {0, 0, 0, 0, 0, 0, 0, 0};
            __builtin_amdgcn_tensor_load_to_lds(g0, g1, g2, g3, g4, 0);
#else
            __builtin_amdgcn_tensor_load_to_lds(g0, g1, g2, g3, 0);
#endif
            __builtin_amdgcn_s_wait_tensorcnt(0);
        }
#else
        // fallback: manual weight staging
#pragma unroll
        for (int j = 0; j < 6; ++j) {
            const int qq = sq0 + j;
            *(v8bf*)&ldsA[srow * CHN + qq * 8] =
                *(const v8bf*)(pArow + t * CHN + qq * 8);
        }
#endif
        __syncthreads();

        // ---- consume tile: 6 k-blocks x (1 A-frag, 2 B-frags, 2 wmma) ----
#pragma unroll
        for (int cb = 0; cb < CHN / 32; ++cb) {
            const __bf16* fa = &ldsA[(mo + nn) * CHN + cb * 32];
            const v8bf alo = *(const v8bf*)(fa + kA0);
            const v8bf ahi = *(const v8bf*)(fa + kA1);
            const v16bf a = __builtin_shufflevector(
                alo, ahi, 0, 1, 2, 3, 4, 5, 6, 7, 8, 9, 10, 11, 12, 13, 14,
                15);

            const __bf16* fb0 = &ldsB[(n0 + nn) * CHN + cb * 32 + hi * 16];
            const v8bf b0lo = *(const v8bf*)(fb0);
            const v8bf b0hi = *(const v8bf*)(fb0 + 8);
            const v16bf b0 = __builtin_shufflevector(
                b0lo, b0hi, 0, 1, 2, 3, 4, 5, 6, 7, 8, 9, 10, 11, 12, 13, 14,
                15);
            acc0 = __builtin_amdgcn_wmma_f32_16x16x32_bf16(
                false, a, false, b0, (short)0, acc0, false, false);

            const __bf16* fb1 =
                &ldsB[(n0 + 16 + nn) * CHN + cb * 32 + hi * 16];
            const v8bf b1lo = *(const v8bf*)(fb1);
            const v8bf b1hi = *(const v8bf*)(fb1 + 8);
            const v16bf b1 = __builtin_shufflevector(
                b1lo, b1hi, 0, 1, 2, 3, 4, 5, 6, 7, 8, 9, 10, 11, 12, 13, 14,
                15);
            acc1 = __builtin_amdgcn_wmma_f32_16x16x32_bf16(
                false, a, false, b1, (short)0, acc1, false, false);
        }
        __syncthreads();                        // protect tile for next tap
    }

    // ---- store D (f32, NCHW): 2 tiles x 8 rows ----
#pragma unroll
    for (int j = 0; j < 2; ++j) {
        const int pj  = p0 + n0 + j * 16 + nn;
        const int img = pj / HW;
        const int rem = pj % HW;
        const v8f acc = j ? acc1 : acc0;
#pragma unroll
        for (int r = 0; r < 8; ++r) {
            const int m = coB + mo + (hi ? 8 : 0) + r;
            out[(img * CHN + m) * HW + rem] = acc[r];
        }
    }
}

// ---------------------------------------------------------------------------
// Memory-bound branches: zeros / identity / avg_pool3 / max_pool3.
// ---------------------------------------------------------------------------
__global__ __launch_bounds__(256) void pointwise_kernel(
    const float* __restrict__ x, const int* __restrict__ idx_ws,
    float* __restrict__ out) {
    const int idx = *idx_ws;
    if (idx == 2 || idx == 3) return;             // conv kernels own d_out
    const int i = blockIdx.x * 256 + threadIdx.x;
    if (i >= NELEM) return;

    if (idx == 0) { out[i] = 0.0f; return; }
    if (idx == 1) { out[i] = x[i]; return; }

    const int pw = i % IMW;
    const int ph = (i / IMW) % IMH;
    const int base = i - ph * IMW - pw;           // start of this (n,c) plane

    if (idx == 4) {                               // avg_pool3: zero-pad, /9
        float s = 0.0f;
#pragma unroll
        for (int dh = -1; dh <= 1; ++dh)
#pragma unroll
            for (int dw = -1; dw <= 1; ++dw) {
                const int h2 = ph + dh, w2 = pw + dw;
                if (h2 >= 0 && h2 < IMH && w2 >= 0 && w2 < IMW)
                    s += x[base + h2 * IMW + w2];
            }
        out[i] = s / 9.0f;
    } else {                                      // max_pool3: -inf pad
        float m = -__builtin_inff();
#pragma unroll
        for (int dh = -1; dh <= 1; ++dh)
#pragma unroll
            for (int dw = -1; dw <= 1; ++dw) {
                const int h2 = ph + dh, w2 = pw + dw;
                if (h2 >= 0 && h2 < IMH && w2 >= 0 && w2 < IMW)
                    m = fmaxf(m, x[base + h2 * IMW + w2]);
            }
        out[i] = m;
    }
}

// ---------------------------------------------------------------------------
extern "C" void kernel_launch(void* const* d_in, const int* in_sizes, int n_in,
                              void* d_out, int out_size, void* d_ws,
                              size_t ws_size, hipStream_t stream) {
    const float* x    = (const float*)d_in[0];
    const float* arch = (const float*)d_in[1];
    const float* w3   = (const float*)d_in[2];
    const float* w5   = (const float*)d_in[3];
    float* out        = (float*)d_out;

    char* ws     = (char*)d_ws;
    __bf16* xb   = (__bf16*)ws;                    // 38.5 MB NHWC relu bf16
    __bf16* wb   = (__bf16*)(ws + XB_BYTES);       // <=1.8 MB [co][t][ci] bf16
    int*    idxp = (int*)(ws + IDX_OFF);           // sampled op

    // 1) sample the architecture gate (device-side, graph-safe)
    pick_op_kernel<<<1, 64, 0, stream>>>(arch, idxp);

    // 2) precision/layout repack for the conv branches (early-exit otherwise)
    const int elem_blocks = (NELEM + 255) / 256;   // 75264
    prep_act_kernel<<<elem_blocks, 256, 0, stream>>>(x, idxp, xb);
    prep_wgt_kernel<9, 2><<<(CHN * CHN * 9 + 255) / 256, 256, 0, stream>>>(
        w3, idxp, wb);
    prep_wgt_kernel<25, 3><<<(CHN * CHN * 25 + 255) / 256, 256, 0, stream>>>(
        w5, idxp, wb);

    // 3) memory-bound branches
    pointwise_kernel<<<elem_blocks, 256, 0, stream>>>(x, idxp, out);

    // 4) WMMA convs: (192/64) x (100352/64) = 4704 blocks of 256 threads
    const int conv_blocks = (CHN / 64) * (NPOS / 64);
    conv_wmma_kernel<3, 3, 2><<<conv_blocks, 256, 0, stream>>>(xb, wb, idxp,
                                                               out);
    conv_wmma_kernel<5, 5, 3><<<conv_blocks, 256, 0, stream>>>(xb, wb, idxp,
                                                               out);
}
